// dual_att_layer_68573447848479
// MI455X (gfx1250) — matrix operand order repaired and checked by reading, not compiled, not run on hardware
//
#include <hip/hip_runtime.h>
#include <hip/hip_bf16.h>
#include <stdint.h>

// MI455X / gfx1250 fused dual-attention layer.
//
// Roofline: adj_mat stream (8192^2 x 4B = 268 MB) dominates -> ~11.5us floor at
// 23.3 TB/s. Structure:
//   * Stage-1 GEMM collapsed algebraically: f1 = inlayer2 @ (W1^T w2) + b2.
//   * Single pass over adj, online (flash) softmax, NT b128 loads,
//     register-pipelined one chunk ahead (branchless wrap).
//   * P@V on the matrix pipe: 2x v_wmma_f32_16x16x32_bf16 per 64-wide chunk.
//   * V staged via CDNA5 async copy (global_load_async_to_lds_b128) from a
//     pre-converted bf16 V^T buffer, double-buffered in LDS: copy for chunk
//     n+1 issued at top of iteration n, drained with s_wait_asynccnt 4
//     (in-order completion per wave), so the copy hides behind a full
//     iteration instead of just the score phase.

typedef __attribute__((ext_vector_type(16))) __bf16 v16bf;
typedef __attribute__((ext_vector_type(8)))  float  v8f;
typedef __attribute__((ext_vector_type(4)))  float  f32x4;   // clang-native vec

#define N_NODES 8192
#define EMB     128
#define E2      256
#define HID     128
#define MT      16    // rows per workgroup (WMMA M)
#define KC      64    // j-chunk (2x WMMA K)
#define VP      72    // sVT pitch (bf16): 144B = 36 dwords -> bank-conflict-free
#define PP      72    // sP pitch (bf16)

// ---------------------------------------------------------------------------
// Kernel 1: u2 = W1^T w2, u3 = W1^T w3   (one 256-thread block)
// ---------------------------------------------------------------------------
__global__ void k_compute_u(const float* __restrict__ W1,
                            const float* __restrict__ w2,
                            const float* __restrict__ w3,
                            float* __restrict__ u2, float* __restrict__ u3) {
    int e = threadIdx.x;                      // 0..255
    float s2 = 0.f, s3 = 0.f;
    for (int h = 0; h < HID; ++h) {
        float w = W1[h * E2 + e];             // coalesced over e
        s2 += w * w2[h];
        s3 += w * w3[h];
    }
    u2[e] = s2;
    u3[e] = s3;
}

// ---------------------------------------------------------------------------
// Kernel 2: f1[i] = inlayer2[i,:].u2 + b2 ; f2[i] = inlayer2[i,:].u3 + b3
// one wave32 per row, 8 waves per block
// ---------------------------------------------------------------------------
__global__ __launch_bounds__(256)
void k_compute_f(const float* __restrict__ in2,
                 const float* __restrict__ u2, const float* __restrict__ u3,
                 const float* __restrict__ pb2, const float* __restrict__ pb3,
                 float* __restrict__ f1, float* __restrict__ f2) {
    const int wave = threadIdx.x >> 5;        // wave32 on gfx1250
    const int lane = threadIdx.x & 31;
    const int row  = blockIdx.x * 8 + wave;
    const float* rp = in2 + (long)row * E2;
    float a2 = 0.f, a3 = 0.f;
#pragma unroll
    for (int k = lane; k < E2; k += 32) {
        float x = rp[k];
        a2 += x * u2[k];
        a3 += x * u3[k];
    }
#pragma unroll
    for (int off = 16; off; off >>= 1) {
        a2 += __shfl_xor(a2, off);
        a3 += __shfl_xor(a3, off);
    }
    if (lane == 0) {
        f1[row] = a2 + pb2[0];
        f2[row] = a3 + pb3[0];
    }
}

// ---------------------------------------------------------------------------
// Kernel 3: VT[c][n] = bf16(inlayer[n][c])  (tiled transpose+convert, 6MB once)
// ---------------------------------------------------------------------------
__global__ __launch_bounds__(256)
void k_cvt_transpose(const float* __restrict__ in, __bf16* __restrict__ vt) {
    __shared__ __bf16 tile[32][33];
    const int n0 = blockIdx.x * 32;
    const int c0 = blockIdx.y * 32;
    const int j  = threadIdx.x & 31;
    const int i  = threadIdx.x >> 5;          // 0..7
#pragma unroll
    for (int it = 0; it < 4; ++it)            // read coalesced along EMB
        tile[i + 8 * it][j] = (__bf16)in[(long)(n0 + i + 8 * it) * EMB + c0 + j];
    __syncthreads();
#pragma unroll
    for (int it = 0; it < 4; ++it)            // write coalesced along N
        vt[(long)(c0 + i + 8 * it) * N_NODES + n0 + j] = tile[j][i + 8 * it];
}

// issue 16KB async V^T chunk copy: VT[c][j0..j0+63] -> LDS buffer at ldsBase
__device__ __forceinline__
void stage_async(unsigned ldsBase, const __bf16* vt, int j0, int tid) {
#pragma unroll
    for (int i = 0; i < 4; ++i) {
        int id = tid + i * 256;               // 0..1023 transfers of 16B
        int c  = id >> 3;
        int q  = id & 7;
        unsigned lds = ldsBase + (unsigned)(c * (VP * 2) + q * 16);
        unsigned gof = (unsigned)(c * N_NODES + j0) * 2u + (unsigned)(q * 16);
        asm volatile("global_load_async_to_lds_b128 %0, %1, %2"
                     :: "v"(lds), "v"(gof), "s"(vt) : "memory");
    }
}

// ---------------------------------------------------------------------------
// Kernel 4: fused masked-softmax attention (flash-style) + bf16 WMMA
// Block = 256 threads = 8 waves, owns 16 rows; wave w owns out cols [16w,16w+16)
// ---------------------------------------------------------------------------
__global__ __launch_bounds__(256)
void k_fused_attn(const float* __restrict__ adj,
                  const float* __restrict__ f1g,
                  const float* __restrict__ f2g,
                  const __bf16* __restrict__ vt,
                  float* __restrict__ out) {
    __shared__ __bf16 sVT[2][EMB][VP];        // double-buffered async V^T chunk
    __shared__ __bf16 sP[MT][PP];             // probs (WMMA A tile)
    __shared__ float  sM[MT], sL[MT], sScale[MT], sF1[MT];

    const int tid  = threadIdx.x;
    const int lane = tid & 31;
    const int wave = tid >> 5;                // 0..7
    const int i0   = blockIdx.x * MT;
    const int col0 = wave * 16;
    const int half = lane >> 4;
    const int lmod = lane & 15;
    const int r    = tid >> 4;                // score phase: row 0..15
    const int c0   = (tid & 15) * 4;          // 4 adj columns per thread

    if (tid < MT) {
        sF1[tid] = f1g[i0 + tid];
        sM[tid]  = -1e30f;
        sL[tid]  = 0.0f;
    }
    v8f acc = {};
    // flat LDS addresses carry the wave-relative offset in addr[31:0]
    const unsigned sVT_base = (unsigned)(uintptr_t)&sVT[0][0][0];
    const unsigned bufBytes = EMB * VP * 2;
    const float* adjRow = adj + (long)(i0 + r) * N_NODES + c0;

    // prologue: stage chunk 0 into buffer 0; prime adj/f2 registers
    stage_async(sVT_base, vt, 0, tid);
    f32x4 av = __builtin_nontemporal_load((const f32x4*)(adjRow));
    f32x4 fv = *(const f32x4*)(f2g + c0);
    int cur = 0;
    __syncthreads();

    for (int j0 = 0; j0 < N_NODES; j0 += KC) {
        // --- 1) issue async stage of NEXT chunk into the other buffer ------
        const int jn = (j0 + KC) & (N_NODES - 1);     // branchless wrap
        stage_async(sVT_base + (unsigned)(cur ^ 1) * bufBytes, vt, jn, tid);
        // prefetch next chunk's adj / f2 into registers (full-iteration slack)
        f32x4 avn = __builtin_nontemporal_load((const f32x4*)(adjRow + jn));
        f32x4 fvn = *(const f32x4*)(f2g + jn + c0);

        // --- 2) scores + online softmax for current chunk ------------------
        const float f1r = sF1[r];
        float sc0, sc1, sc2, sc3;
        {
            float z, lr;
            z = av.x * (f1r + fv.x); lr = z > 0.f ? z : 0.01f * z; sc0 = av.x > 0.f ? lr : -1e9f;
            z = av.y * (f1r + fv.y); lr = z > 0.f ? z : 0.01f * z; sc1 = av.y > 0.f ? lr : -1e9f;
            z = av.z * (f1r + fv.z); lr = z > 0.f ? z : 0.01f * z; sc2 = av.z > 0.f ? lr : -1e9f;
            z = av.w * (f1r + fv.w); lr = z > 0.f ? z : 0.01f * z; sc3 = av.w > 0.f ? lr : -1e9f;
        }
        float cmax = fmaxf(fmaxf(sc0, sc1), fmaxf(sc2, sc3));
#pragma unroll
        for (int m = 8; m; m >>= 1)           // 16-lane row-group reduction
            cmax = fmaxf(cmax, __shfl_xor(cmax, m));
        const float mold = sM[r];
        const float mnew = fmaxf(mold, cmax);
        float p0 = __expf(sc0 - mnew), p1 = __expf(sc1 - mnew);
        float p2 = __expf(sc2 - mnew), p3 = __expf(sc3 - mnew);
        float ps = (p0 + p1) + (p2 + p3);
#pragma unroll
        for (int m = 8; m; m >>= 1)
            ps += __shfl_xor(ps, m);
        union { __bf16 h[4]; unsigned long long u; } pk;
        pk.h[0] = (__bf16)p0; pk.h[1] = (__bf16)p1;
        pk.h[2] = (__bf16)p2; pk.h[3] = (__bf16)p3;
        *(unsigned long long*)&sP[r][c0] = pk.u;   // 8B aligned, single ds store
        if ((tid & 15) == 0) {
            float scl  = __expf(mold - mnew);
            sScale[r]  = scl;
            sM[r]      = mnew;
            sL[r]      = sL[r] * scl + ps;
        }

        // current buffer's 4 transfers (issued last iteration) are the oldest;
        // async loads complete in order per wave -> <=4 outstanding drains them
        asm volatile("s_wait_asynccnt 0x4" ::: "memory");
        __syncthreads();

        // --- 3) per-wave WMMA: acc = acc*scale + P(16x64) @ V(64x16) -------
        const __bf16* prow = &sP[lmod][0];
        const __bf16* vrow = &sVT[cur][col0 + lmod][0];
        v16bf a0, a1, b0, b1;
#pragma unroll
        for (int v = 0; v < 8; ++v) {
            // A 16x32 bf16 layout (ISA 7.12.2): VGPR v<4 -> K0..15, v>=4 ->
            // K16..31, lane half selects 8-wide subrange, 2 K packed per VGPR.
            int ka = ((v >> 2) * 16) + half * 8 + 2 * (v & 3);
            a0[2 * v]     = prow[ka];
            a0[2 * v + 1] = prow[ka + 1];
            a1[2 * v]     = prow[ka + 32];
            a1[2 * v + 1] = prow[ka + 33];
            // B 32x16 bf16: same K layout as A (lane half selects 8-wide subrange).
            int kb = ((v >> 2) * 16) + half * 8 + 2 * (v & 3);
            b0[2 * v]     = vrow[kb];
            b0[2 * v + 1] = vrow[kb + 1];
            b1[2 * v]     = vrow[kb + 32];
            b1[2 * v + 1] = vrow[kb + 33];
        }
#pragma unroll
        for (int t = 0; t < 8; ++t)           // VGPR t holds row t + 8*half
            acc[t] *= sScale[t + 8 * half];
        acc = __builtin_amdgcn_wmma_f32_16x16x32_bf16(
                  false, a0, false, b0, (short)0, acc, false, false);
        acc = __builtin_amdgcn_wmma_f32_16x16x32_bf16(
                  false, a1, false, b1, (short)0, acc, false, false);
        __syncthreads();                      // protect sP/sScale/buffer reuse

        av = avn; fv = fvn; cur ^= 1;
    }

    // --- epilogue: divide by softmax denominator, store ---------------------
#pragma unroll
    for (int t = 0; t < 8; ++t) {
        int m = t + 8 * half;
        out[(long)(i0 + m) * EMB + col0 + lmod] = acc[t] / sL[m];
    }
}

// ---------------------------------------------------------------------------
extern "C" void kernel_launch(void* const* d_in, const int* in_sizes, int n_in,
                              void* d_out, int out_size, void* d_ws, size_t ws_size,
                              hipStream_t stream) {
    const float* inlayer  = (const float*)d_in[0];
    const float* inlayer2 = (const float*)d_in[1];
    const float* adj      = (const float*)d_in[2];
    const float* W1       = (const float*)d_in[3];
    const float* w2       = (const float*)d_in[4];
    const float* b2       = (const float*)d_in[5];
    const float* w3       = (const float*)d_in[6];
    const float* b3       = (const float*)d_in[7];
    float* out = (float*)d_out;

    // workspace: VT bf16 [128][8192] (2MB) then f32 region (66KB)
    __bf16* vt = (__bf16*)d_ws;
    float* fws = (float*)((char*)d_ws + (size_t)EMB * N_NODES * sizeof(__bf16));
    float* u2 = fws;
    float* u3 = fws + E2;
    float* f1 = fws + 2 * E2;
    float* f2 = f1 + N_NODES;

    k_compute_u<<<1, 256, 0, stream>>>(W1, w2, w3, u2, u3);
    k_compute_f<<<N_NODES / 8, 256, 0, stream>>>(inlayer2, u2, u3, b2, b3, f1, f2);
    k_cvt_transpose<<<dim3(N_NODES / 32, EMB / 32), 256, 0, stream>>>(inlayer, vt);
    k_fused_attn<<<N_NODES / MT, 256, 0, stream>>>(adj, f1, f2, vt, out);
}
